// DistributionalOrderedEncoder_12610023981510
// MI455X (gfx1250) — matrix operand order repaired and checked by reading, not compiled, not run on hardware
//
#include <hip/hip_runtime.h>
#include <hip/hip_bf16.h>

typedef __attribute__((ext_vector_type(16))) _Float16 v16h;
typedef __attribute__((ext_vector_type(8)))  _Float16 v8h;
typedef __attribute__((ext_vector_type(8)))  float    v8f;
typedef __attribute__((ext_vector_type(4)))  int      v4i;

typedef __attribute__((address_space(1))) v4i* gptr_b128;   // global int4*
typedef __attribute__((address_space(3))) v4i* lptr_b128;   // LDS int4*

#define BB 128
#define TT 64
#define SS 64
#define DD 64
#define HH 64
#define NEGF (-1000000000.0f)

// ---------------------------------------------------------------------------
// Async global->LDS copy of one 16B chunk (GLOBAL_LOAD_ASYNC_TO_LDS_B128).
// Builtin signature: void(v4i AS1* src, v4i AS3* dst, imm offset, imm cpol)
// ---------------------------------------------------------------------------
#if __has_builtin(__builtin_amdgcn_global_load_async_to_lds_b128)
#define HAVE_ASYNC_LDS 1
#else
#define HAVE_ASYNC_LDS 0
#endif

__device__ __forceinline__ void copy16_to_lds(void* lds_dst, const void* gsrc) {
#if HAVE_ASYNC_LDS
    __builtin_amdgcn_global_load_async_to_lds_b128(
        (gptr_b128)(gsrc), (lptr_b128)(lds_dst), 0, 0);
#else
    *(float4*)lds_dst = *(const float4*)gsrc;
#endif
}
__device__ __forceinline__ void async_copy_wait() {
#if HAVE_ASYNC_LDS
    asm volatile("s_wait_asynccnt 0" ::: "memory");
#endif
}

// Concatenate two 8-half chunks into a WMMA 16-half fragment (register-only).
__device__ __forceinline__ v16h cat8(v8h lo, v8h hi) {
    return __builtin_shufflevector(lo, hi, 0, 1, 2, 3, 4, 5, 6, 7,
                                           8, 9, 10, 11, 12, 13, 14, 15);
}
// Convert 8 f32 (two float4) to 8 f16.
__device__ __forceinline__ v8h cvt8(float4 x, float4 y) {
    v8h r;
    r[0] = (_Float16)x.x; r[1] = (_Float16)x.y;
    r[2] = (_Float16)x.z; r[3] = (_Float16)x.w;
    r[4] = (_Float16)y.x; r[5] = (_Float16)y.y;
    r[6] = (_Float16)y.z; r[7] = (_Float16)y.w;
    return r;
}

// ---------------------------------------------------------------------------
// Pre-kernel (runs once, tiny):
//   pe16[s,h]   = f16( relu(pos @ pos_w1 + pos_b1) @ pos_w2 + pos_b2 )
//   w16t[n,k]   = f16( att_w1[k,n] ), k in [0,128)   (transposed [w_emb;w_pos])
//   cbias[b,n]  = att_b1[n] + target_emb[b,:] @ w_tgt[:,n]
// ---------------------------------------------------------------------------
__global__ __launch_bounds__(64) void doe_precompute(
    const float* __restrict__ target_emb,
    const float* __restrict__ pos_w1, const float* __restrict__ pos_b1,
    const float* __restrict__ pos_w2, const float* __restrict__ pos_b2,
    const float* __restrict__ att_w1, const float* __restrict__ att_b1,
    _Float16* __restrict__ pe16, _Float16* __restrict__ w16t,
    float* __restrict__ cbias)
{
    if (blockIdx.x == 0) {
        int s = threadIdx.x;
        if (s < SS) {
            float p = (float)s / (float)(SS - 1);
            float l1[HH];
            for (int h = 0; h < HH; ++h)
                l1[h] = fmaxf(p * pos_w1[h] + pos_b1[h], 0.0f);
            for (int h = 0; h < HH; ++h) {
                float acc = pos_b2[h];
                for (int k = 0; k < HH; ++k) acc += l1[k] * pos_w2[k * HH + h];
                pe16[s * HH + h] = (_Float16)acc;
            }
        }
        for (int i = threadIdx.x; i < 2 * DD * HH; i += 64) {
            const int k = i >> 6, n = i & 63;          // i = k*64 + n
            w16t[n * 128 + k] = (_Float16)att_w1[i];   // transposed store
        }
    } else {
        int b = blockIdx.x - 1;
        int n = threadIdx.x;
        if (n < HH) {
            float acc = att_b1[n];
            for (int d = 0; d < DD; ++d)
                acc += target_emb[b * DD + d] * att_w1[(DD + HH + d) * HH + n];
            cbias[b * HH + n] = acc;
        }
    }
}

// ---------------------------------------------------------------------------
// Main fused kernel: one workgroup (128 threads = 4 waves) per (b,t).
//   logits  : [X | pos_enc] (64x128 f16) @ [w_emb ; w_pos] (128x64 f16)
//             4 chained v_wmma_f32_16x16x32_f16 per 16-col tile,
//             accumulator seeded with cbias[b,n];
//   fused relu + att_w2 matvec (ds_add_f32), masked softmax,
//   attn-weighted sum over X (f32), projection, LayerNorm.
// ---------------------------------------------------------------------------
__global__ __launch_bounds__(128) void doe_main(
    const float*    __restrict__ seg_emb,   // (B,T,S,D) f32
    const int*      __restrict__ seg_mask,  // (B,T,S)
    const _Float16* __restrict__ pe16,      // (S,H) f16
    const _Float16* __restrict__ w16t,      // (H, 2D) f16, transposed weights
    const float*    __restrict__ cbias,     // (B,H) = b1 + tgt bias
    const float*    __restrict__ att_w2,    // (H,1)
    const float*    __restrict__ proj_w,    // (D,D)
    const float*    __restrict__ proj_b,    // (D)
    const float*    __restrict__ ln_gamma, const float* __restrict__ ln_beta,
    float*          __restrict__ out)       // (B,T,D)
{
    // Row strides are 16B multiples: aligned B128 chunks + bank spreading.
    __shared__ __align__(16) float    sX [SS][68];    // X tile f32   (17.4 KB)
    __shared__ __align__(16) _Float16 sPE[SS][72];    // pos_enc f16  ( 9.2 KB)
    __shared__ __align__(16) _Float16 sWt[HH][136];   // W^T f16      (17.4 KB)
    __shared__ float sScore[SS];
    __shared__ float sAttn[SS];
    __shared__ float sRed[SS];
    __shared__ float sSeg[DD];
    __shared__ float sR1[DD];
    __shared__ float sR2[DD];

    const int bt  = blockIdx.x;
    const int b   = bt / TT;
    const int tid = threadIdx.x;

    // --- async-stage X (16KB), pos_enc (8KB), W^T (16KB) into LDS ---
    const float* Xg = seg_emb + (size_t)bt * (SS * DD);
    for (int i = tid; i < SS * DD / 4; i += 128)              // 16B f32 chunks
        copy16_to_lds(&sX[i >> 4][(i & 15) * 4], Xg + i * 4);
    for (int i = tid; i < SS * HH / 8; i += 128)              // 16B f16 chunks
        copy16_to_lds(&sPE[i >> 3][(i & 7) * 8], pe16 + i * 8);
    for (int i = tid; i < HH * 128 / 8; i += 128)             // rows of 128 halves
        copy16_to_lds(&sWt[i >> 4][(i & 15) * 8], w16t + i * 8);
    if (tid < SS) sScore[tid] = 0.0f;
    async_copy_wait();
    __syncthreads();

    const int wave = tid >> 5;             // 0..3 : owns rows [16w, 16w+16)
    const int lane = tid & 31;
    const int mrow = lane & 15;
    const int half = lane >> 4;            // 0 | 1

    // --- A fragments (ISA 16-bit A layout): lane = row M, runs of 8 in K ---
    const int Mg = wave * 16 + mrow;
    const int q  = 2 * half;               // kb/4 with kb = 8*half
    const float4* px = (const float4*)&sX[Mg][0];    // 8-float runs, f32 -> f16
    const v8h*    pp = (const v8h*)  &sPE[Mg][0];    // 8-half runs, direct
    v16h a0 = cat8(cvt8(px[q],     px[q + 1]),  cvt8(px[q + 4],  px[q + 5]));
    v16h a1 = cat8(cvt8(px[q + 8], px[q + 9]),  cvt8(px[q + 12], px[q + 13]));
    v16h a2 = cat8(pp[half],     pp[half + 2]);
    v16h a3 = cat8(pp[half + 4], pp[half + 6]);

    // --- 4 column tiles: 4 chained WMMAs (K=128), bias-seeded accumulator ---
    #pragma unroll
    for (int nt = 0; nt < 4; ++nt) {
        const int n = nt * 16 + mrow;      // lane's output column
        const v8h* pb = (const v8h*)&sWt[n][0];      // 8-half runs per frag
        const v16h b0 = cat8(pb[half],      pb[half + 2]);
        const v16h b1 = cat8(pb[half + 4],  pb[half + 6]);
        const v16h b2 = cat8(pb[half + 8],  pb[half + 10]);
        const v16h b3 = cat8(pb[half + 12], pb[half + 14]);
        const float cb = cbias[b * HH + n];
        v8f acc;
        #pragma unroll
        for (int r = 0; r < 8; ++r) acc[r] = cb;
        acc = __builtin_amdgcn_wmma_f32_16x16x32_f16(false, a0, false, b0,
                                                     (short)0, acc, false, false);
        acc = __builtin_amdgcn_wmma_f32_16x16x32_f16(false, a1, false, b1,
                                                     (short)0, acc, false, false);
        acc = __builtin_amdgcn_wmma_f32_16x16x32_f16(false, a2, false, b2,
                                                     (short)0, acc, false, false);
        acc = __builtin_amdgcn_wmma_f32_16x16x32_f16(false, a3, false, b3,
                                                     (short)0, acc, false, false);
        const float w2n = att_w2[n];
        #pragma unroll
        for (int r = 0; r < 8; ++r) {
            const int s = wave * 16 + half * 8 + r;     // C/D layout row
            atomicAdd(&sScore[s], fmaxf(acc[r], 0.0f) * w2n);  // ds_add_f32
        }
    }
    __syncthreads();

    // --- masked softmax over S (att_b2 dropped: softmax shift-invariant) ---
    float sc = NEGF;
    if (tid < SS) {
        sc = (seg_mask[(size_t)bt * SS + tid] == 0) ? NEGF : sScore[tid];
        sRed[tid] = sc;
    }
    __syncthreads();
    for (int off = 32; off > 0; off >>= 1) {
        if (tid < off) sRed[tid] = fmaxf(sRed[tid], sRed[tid + off]);
        __syncthreads();
    }
    const float mx = sRed[0];
    __syncthreads();
    float ev = 0.0f;
    if (tid < SS) { ev = __expf(sc - mx); sRed[tid] = ev; }
    __syncthreads();
    for (int off = 32; off > 0; off >>= 1) {
        if (tid < off) sRed[tid] += sRed[tid + off];
        __syncthreads();
    }
    const float esum = sRed[0];
    __syncthreads();
    if (tid < SS) sAttn[tid] = ev / esum;
    __syncthreads();

    // --- seg[d] = sum_s attn[s] * X[s,d]  (f32, from LDS) ---
    const int d  = tid & 63;
    const int sh = tid >> 6;
    float part = 0.0f;
    for (int s = sh * 32; s < sh * 32 + 32; ++s) part += sAttn[s] * sX[s][d];
    if (sh == 1) sRed[d] = part;
    __syncthreads();
    if (sh == 0) sSeg[d] = part + sRed[d];
    __syncthreads();

    // --- projection + LayerNorm (threads 0..63, one output element each) ---
    float y = 0.0f;
    if (tid < DD) {
        for (int k = 0; k < DD; ++k) y += sSeg[k] * proj_w[k * DD + d];
        y += proj_b[d];
        sR1[tid] = y;
        sR2[tid] = y * y;
    }
    __syncthreads();
    for (int off = 32; off > 0; off >>= 1) {
        if (tid < off) { sR1[tid] += sR1[tid + off]; sR2[tid] += sR2[tid + off]; }
        __syncthreads();
    }
    if (tid < DD) {
        const float mu  = sR1[0] * (1.0f / DD);
        const float var = sR2[0] * (1.0f / DD) - mu * mu;
        const float nrm = (y - mu) * __frsqrt_rn(var + 1e-5f);
        out[(size_t)bt * DD + tid] = nrm * ln_gamma[tid] + ln_beta[tid];
    }
}

extern "C" void kernel_launch(void* const* d_in, const int* in_sizes, int n_in,
                              void* d_out, int out_size, void* d_ws, size_t ws_size,
                              hipStream_t stream) {
    const float* seg_emb    = (const float*)d_in[0];
    const int*   seg_mask   = (const int*)  d_in[1];
    const float* target_emb = (const float*)d_in[2];
    const float* pos_w1     = (const float*)d_in[3];
    const float* pos_b1     = (const float*)d_in[4];
    const float* pos_w2     = (const float*)d_in[5];
    const float* pos_b2     = (const float*)d_in[6];
    const float* att_w1     = (const float*)d_in[7];
    const float* att_b1     = (const float*)d_in[8];
    const float* att_w2     = (const float*)d_in[9];
    // d_in[10] = att_b2 : unused (softmax is shift-invariant)
    const float* proj_w     = (const float*)d_in[11];
    const float* proj_b     = (const float*)d_in[12];
    const float* ln_gamma   = (const float*)d_in[13];
    const float* ln_beta    = (const float*)d_in[14];
    float*       out        = (float*)d_out;

    // Workspace layout (16B-aligned sections):
    char* ws = (char*)d_ws;
    _Float16* pe16  = (_Float16*)(ws);                  //  8 KB : (S,H) f16
    _Float16* w16t  = (_Float16*)(ws + 8 * 1024);       // 16 KB : (H,2D) f16 (transposed)
    float*    cbias = (float*)   (ws + 24 * 1024);      // 32 KB : (B,H) f32

    doe_precompute<<<BB + 1, 64, 0, stream>>>(target_emb, pos_w1, pos_b1,
                                              pos_w2, pos_b2, att_w1, att_b1,
                                              pe16, w16t, cbias);
    doe_main<<<BB * TT, 128, 0, stream>>>(seg_emb, seg_mask, pe16, w16t, cbias,
                                          att_w2, proj_w, proj_b,
                                          ln_gamma, ln_beta, out);
}
